// Decoder_83872121356943
// MI455X (gfx1250) — compile-verified, matrix operand order checked by reading, not run on hardware
//
#include <hip/hip_runtime.h>

// ---------------------------------------------------------------------------
// Shapes (compile-time constants from the reference)
// ---------------------------------------------------------------------------
#define LNUM 12
#define HNUM 12
#define CDIM 768
#define HD   64
#define TQ   512
#define BNUM 8
#define NTOK (BNUM * TQ)   // 4096 rows
#define NVH  128
#define VOC  5

typedef unsigned short u16;
typedef unsigned int   u32;
typedef __attribute__((ext_vector_type(16))) __bf16 v16bf;
typedef __attribute__((ext_vector_type(8)))  float  v8f;
typedef __attribute__((ext_vector_type(4)))  u32    u32x4;
// Exact pointee type of the async-LDS builtin's pointer params (per hipcc
// diagnostic: "__attribute__((__vector_size__(4 * sizeof(int)))) int *").
typedef int v4i __attribute__((vector_size(16)));

#if defined(__has_builtin)
#if __has_builtin(__builtin_amdgcn_global_load_async_to_lds_b128)
#define USE_ASYNC_LDS 1
#endif
#endif

#if USE_ASYNC_LDS
__device__ __forceinline__ void async_b128(const u16* g, u16* l) {
  __builtin_amdgcn_global_load_async_to_lds_b128(
      (__attribute__((address_space(1))) v4i*)(u16*)g,
      (__attribute__((address_space(3))) v4i*)l, 0, 0);
}
#endif

// ---------------------------------------------------------------------------
// bf16 helpers (round-to-nearest-even)
// ---------------------------------------------------------------------------
__device__ __forceinline__ u16 f2bf(float f) {
  u32 b = __builtin_bit_cast(u32, f);
  return (u16)((b + 0x7FFFu + ((b >> 16) & 1u)) >> 16);
}
__device__ __forceinline__ float bf2f(u16 h) {
  u32 b = ((u32)h) << 16;
  return __builtin_bit_cast(float, b);
}
__device__ __forceinline__ float gelu_f(float v) {
  return 0.5f * v * (1.0f + tanhf(0.7978845608028654f * (v + 0.044715f * v * v * v)));
}
__device__ __forceinline__ float blockReduceSum(float v, float* red) {
  int tid = threadIdx.x;
  red[tid] = v; __syncthreads();
  for (int off = 128; off > 0; off >>= 1) {
    if (tid < off) red[tid] += red[tid + off];
    __syncthreads();
  }
  float r = red[0]; __syncthreads();
  return r;
}
__device__ __forceinline__ float blockReduceMax(float v, float* red) {
  int tid = threadIdx.x;
  red[tid] = v; __syncthreads();
  for (int off = 128; off > 0; off >>= 1) {
    if (tid < off) red[tid] = fmaxf(red[tid], red[tid + off]);
    __syncthreads();
  }
  float r = red[0]; __syncthreads();
  return r;
}

// ---------------------------------------------------------------------------
// Embedding: h[b*T+t][c] = wte[x[b,t]][c] + wpe[t][c]
// ---------------------------------------------------------------------------
__global__ void embed_kernel(const int* __restrict__ x,
                             const float* __restrict__ wte,
                             const float* __restrict__ wpe,
                             float* __restrict__ h) {
  int row = blockIdx.x;
  int t   = row & (TQ - 1);
  int tok = x[row];
  const float* we = wte + (size_t)tok * CDIM;
  const float* pe = wpe + (size_t)t * CDIM;
  float* out = h + (size_t)row * CDIM;
  for (int c = threadIdx.x; c < CDIM; c += blockDim.x) out[c] = we[c] + pe[c];
}

// ---------------------------------------------------------------------------
// LayerNorm over C=768 -> bf16 output (block = 256, one row per block)
// ---------------------------------------------------------------------------
__global__ __launch_bounds__(256) void ln_bf16_kernel(const float* __restrict__ in,
                                                      const float* __restrict__ s,
                                                      const float* __restrict__ b,
                                                      u16* __restrict__ out) {
  __shared__ float red[256];
  int row = blockIdx.x;
  int tid = threadIdx.x;
  const float* xr = in + (size_t)row * CDIM;
  float vals[3];
  float sum = 0.f;
#pragma unroll
  for (int i = 0; i < 3; ++i) { vals[i] = xr[tid + i * 256]; sum += vals[i]; }
  float mean = blockReduceSum(sum, red) * (1.0f / CDIM);
  float sq = 0.f;
#pragma unroll
  for (int i = 0; i < 3; ++i) { float d = vals[i] - mean; sq += d * d; }
  float var = blockReduceSum(sq, red) * (1.0f / CDIM);
  float rstd = rsqrtf(var + 1e-6f);
  u16* orow = out + (size_t)row * CDIM;
#pragma unroll
  for (int i = 0; i < 3; ++i) {
    int c = tid + i * 256;
    orow[c] = f2bf((vals[i] - mean) * rstd * s[c] + b[c]);
  }
}

// ---------------------------------------------------------------------------
// Elementwise conversions
// ---------------------------------------------------------------------------
__global__ void f32_to_bf16_kernel(const float* __restrict__ in, u16* __restrict__ out, long n) {
  long i = (long)blockIdx.x * blockDim.x + threadIdx.x;
  long stride = (long)gridDim.x * blockDim.x;
  for (; i < n; i += stride) out[i] = f2bf(in[i]);
}
__global__ void gelu_bf16_kernel(const float* __restrict__ in, u16* __restrict__ out, long n) {
  long i = (long)blockIdx.x * blockDim.x + threadIdx.x;
  long stride = (long)gridDim.x * blockDim.x;
  for (; i < n; i += stride) out[i] = f2bf(gelu_f(in[i]));
}
__global__ void add_inplace_kernel(float* __restrict__ a, const float* __restrict__ b, long n) {
  long i = (long)blockIdx.x * blockDim.x + threadIdx.x;
  long stride = (long)gridDim.x * blockDim.x;
  for (; i < n; i += stride) a[i] += b[i];
}

// ---------------------------------------------------------------------------
// Batched tiled transpose + f32->bf16:  out[z][c][r] = in[bb,hh][r][c]
//   in element [r][c] at in + bb*sIb + hh*sIh + r*ldi + c   (f32)
//   out is z-major dense: out + z*rows*cols, row stride = rows (bf16)
// Used to pre-transpose all GEMM B operands so the WMMA kernel always
// reads B contiguous-in-K (enables b128 staging with zero branches).
// ---------------------------------------------------------------------------
__global__ __launch_bounds__(256) void transpose_f32_bf16_kernel(
    const float* __restrict__ in, u16* __restrict__ out,
    int rows, int cols, int ldi, long sIb, long sIh, int Hdim) {
  __shared__ float tile[32][33];
  int z = blockIdx.z;
  int bb = z / Hdim, hh = z % Hdim;
  const float* ip = in + bb * sIb + hh * sIh;
  u16* op = out + (long)z * rows * cols;
  int c0 = blockIdx.x * 32, r0 = blockIdx.y * 32;
  int tx = threadIdx.x & 31, ty = threadIdx.x >> 5;   // ty = 0..7
#pragma unroll
  for (int j = 0; j < 4; ++j)
    tile[ty + 8 * j][tx] = ip[(size_t)(r0 + ty + 8 * j) * ldi + c0 + tx];
  __syncthreads();
#pragma unroll
  for (int j = 0; j < 4; ++j)
    op[(size_t)(c0 + ty + 8 * j) * rows + r0 + tx] = f2bf(tile[tx][ty + 8 * j]);
}

// ---------------------------------------------------------------------------
// Softmax over 512-wide rows -> bf16 probabilities
// ---------------------------------------------------------------------------
__global__ __launch_bounds__(256) void softmax_bf16_kernel(const float* __restrict__ in,
                                                           u16* __restrict__ out) {
  __shared__ float red[256];
  long row = blockIdx.x;
  int tid = threadIdx.x;
  const float* xr = in + row * 512;
  u16* orow = out + row * 512;
  float v0 = xr[tid], v1 = xr[tid + 256];
  float m = blockReduceMax(fmaxf(v0, v1), red);
  float e0 = __expf(v0 - m), e1 = __expf(v1 - m);
  float inv = 1.0f / blockReduceSum(e0 + e1, red);
  orow[tid]       = f2bf(e0 * inv);
  orow[tid + 256] = f2bf(e1 * inv);
}

// ---------------------------------------------------------------------------
// Generic batched bf16 WMMA GEMM:  C[z] = alpha * A[z] @ B[z] + bias
//   A: M x K bf16, row stride lda (contiguous in k)
//   B: stored [n][k] bf16, row stride ldb (contiguous in k)  [pre-transposed]
//   C: M x N f32, row stride ldc
//   z = (bb = z/Hdim, hh = z%Hdim) with independent strides.
// Shapes in this model guarantee: M % 64 == 0 (grid-exact), K % 32 == 0,
// all strides/offsets multiples of 8 halves -> every staging access is an
// aligned b128, no bounds checks needed (N=64 case covered by slab slack).
// 256 threads = 8 waves; 64x128 tile; wave = 32x32 = 2x2 wmma accumulators.
// Async path: double-buffered LDS fed by global_load_async_to_lds_b128.
// ---------------------------------------------------------------------------
#define TM 64
#define TN 128
#define TK 32

__global__ __launch_bounds__(256) void gemm_wmma_kernel(
    const u16* __restrict__ Abase, const u16* __restrict__ Bbase,
    const float* __restrict__ bias, float* __restrict__ Cbase,
    int M, int N, int K, int lda, int ldb, int ldc,
    long sAb, long sAh, long sBb, long sBh, long sCb, long sCh,
    int Hdim, float alpha) {
#if USE_ASYNC_LDS
  __shared__ u16 As[2][TM][TK];
  __shared__ u16 Bs[2][TN][TK];
#else
  __shared__ u16 As[1][TM][TK];
  __shared__ u16 Bs[1][TN][TK];
#endif

  int z  = blockIdx.z;
  int bb = z / Hdim, hh = z % Hdim;
  const u16* A = Abase + bb * sAb + hh * sAh;
  const u16* B = Bbase + bb * sBb + hh * sBh;
  float*     C = Cbase + bb * sCb + hh * sCh;

  int m0 = blockIdx.y * TM;
  int n0 = blockIdx.x * TN;
  int tid  = threadIdx.x;
  int lane = tid & 31, wave = tid >> 5;
  int wm = (wave >> 2) * 32;
  int wn = (wave & 3) * 32;
  int idx = lane & 15, half = lane >> 4;

  // staging coordinates: A = one b128 per thread, B = two b128s per thread
  int ar = tid >> 2, ac = (tid & 3) * 8;   // A: 64 rows x 32 k
  int br = tid >> 1, bc = (tid & 1) * 16;  // B: 128 rows x 32 k
  const u16* Ag = A + (size_t)(m0 + ar) * lda + ac;
  const u16* Bg = B + (size_t)(n0 + br) * ldb + bc;

  v8f acc[2][2] = {};
  int nk = K / TK;

  union FragU { v16bf v; u32 u[8]; };

#if USE_ASYNC_LDS
  // -------- async-LDS double-buffered pipeline (CDNA5 path) --------
  async_b128(Ag,     &As[0][ar][ac]);
  async_b128(Bg,     &Bs[0][br][bc]);
  async_b128(Bg + 8, &Bs[0][br][bc + 8]);
  for (int s = 0; s < nk; ++s) {
    int buf = s & 1;
#if __has_builtin(__builtin_amdgcn_s_wait_asynccnt)
    __builtin_amdgcn_s_wait_asynccnt(0);
#else
    asm volatile("s_wait_asynccnt 0x0" ::: "memory");
#endif
    __syncthreads();
    if (s + 1 < nk) {
      int koff = (s + 1) * TK;
      int nbuf = buf ^ 1;
      async_b128(Ag + koff,     &As[nbuf][ar][ac]);
      async_b128(Bg + koff,     &Bs[nbuf][br][bc]);
      async_b128(Bg + koff + 8, &Bs[nbuf][br][bc + 8]);
    }
    FragU af[2], bfr[2];
#pragma unroll
    for (int mi = 0; mi < 2; ++mi) {
      const u32* lo = (const u32*)&As[buf][wm + mi * 16 + idx][half * 8];
      const u32* hi = (const u32*)&As[buf][wm + mi * 16 + idx][16 + half * 8];
#pragma unroll
      for (int q = 0; q < 4; ++q) { af[mi].u[q] = lo[q]; af[mi].u[4 + q] = hi[q]; }
    }
#pragma unroll
    for (int ni = 0; ni < 2; ++ni) {
      const u32* lo = (const u32*)&Bs[buf][wn + ni * 16 + idx][half * 8];
      const u32* hi = (const u32*)&Bs[buf][wn + ni * 16 + idx][16 + half * 8];
#pragma unroll
      for (int q = 0; q < 4; ++q) { bfr[ni].u[q] = lo[q]; bfr[ni].u[4 + q] = hi[q]; }
    }
#pragma unroll
    for (int mi = 0; mi < 2; ++mi)
#pragma unroll
      for (int ni = 0; ni < 2; ++ni)
        acc[mi][ni] = __builtin_amdgcn_wmma_f32_16x16x32_bf16(
            false, af[mi].v, false, bfr[ni].v, (short)0, acc[mi][ni],
            false, false);
  }
#else
  // -------- fallback: vectorized load -> ds_store_b128 staging --------
  for (int s = 0; s < nk; ++s) {
    u32x4 av  = *(const u32x4*)(Ag + s * TK);
    u32x4 bv0 = *(const u32x4*)(Bg + s * TK);
    u32x4 bv1 = *(const u32x4*)(Bg + s * TK + 8);
    if (s + 1 < nk && lane == 0)
      __builtin_prefetch(Ag + (s + 1) * TK, 0, 1);
    __syncthreads();
    *(u32x4*)&As[0][ar][ac]     = av;
    *(u32x4*)&Bs[0][br][bc]     = bv0;
    *(u32x4*)&Bs[0][br][bc + 8] = bv1;
    __syncthreads();
    FragU af[2], bfr[2];
#pragma unroll
    for (int mi = 0; mi < 2; ++mi) {
      const u32* lo = (const u32*)&As[0][wm + mi * 16 + idx][half * 8];
      const u32* hi = (const u32*)&As[0][wm + mi * 16 + idx][16 + half * 8];
#pragma unroll
      for (int q = 0; q < 4; ++q) { af[mi].u[q] = lo[q]; af[mi].u[4 + q] = hi[q]; }
    }
#pragma unroll
    for (int ni = 0; ni < 2; ++ni) {
      const u32* lo = (const u32*)&Bs[0][wn + ni * 16 + idx][half * 8];
      const u32* hi = (const u32*)&Bs[0][wn + ni * 16 + idx][16 + half * 8];
#pragma unroll
      for (int q = 0; q < 4; ++q) { bfr[ni].u[q] = lo[q]; bfr[ni].u[4 + q] = hi[q]; }
    }
#pragma unroll
    for (int mi = 0; mi < 2; ++mi)
#pragma unroll
      for (int ni = 0; ni < 2; ++ni)
        acc[mi][ni] = __builtin_amdgcn_wmma_f32_16x16x32_bf16(
            false, af[mi].v, false, bfr[ni].v, (short)0, acc[mi][ni],
            false, false);
  }
#endif

  // store: lane(idx,half): vgpr r -> M = half*8 + r, N = idx
#pragma unroll
  for (int mi = 0; mi < 2; ++mi)
#pragma unroll
    for (int ni = 0; ni < 2; ++ni)
#pragma unroll
      for (int r = 0; r < 8; ++r) {
        int gm = m0 + wm + mi * 16 + half * 8 + r;
        int gn = n0 + wn + ni * 16 + idx;
        if (gm < M && gn < N) {
          float v = acc[mi][ni][r] * alpha + (bias ? bias[gn] : 0.0f);
          C[(size_t)gm * ldc + gn] = v;
        }
      }
}

// ---------------------------------------------------------------------------
// Policy head: logits[row][v] = dot(hn_bf[row], wte[v]),  V = 5
// ---------------------------------------------------------------------------
__global__ __launch_bounds__(256) void policy_kernel(const u16* __restrict__ hn,
                                                     const float* __restrict__ wte,
                                                     float* __restrict__ out) {
  __shared__ float red[256];
  int row = blockIdx.x;
  int tid = threadIdx.x;
  const u16* xr = hn + (size_t)row * CDIM;
  float xs[3];
#pragma unroll
  for (int i = 0; i < 3; ++i) xs[i] = bf2f(xr[tid + i * 256]);
  for (int v = 0; v < VOC; ++v) {
    const float* w = wte + (size_t)v * CDIM;
    float p = 0.f;
#pragma unroll
    for (int i = 0; i < 3; ++i) p += xs[i] * w[tid + i * 256];
    float tot = blockReduceSum(p, red);
    if (tid == 0) out[(size_t)row * VOC + v] = tot;
  }
}

// ---------------------------------------------------------------------------
// Value head MLP (8 rows, 768->128->128->128->1), single block of 128
// ---------------------------------------------------------------------------
__global__ __launch_bounds__(128) void value_head_kernel(
    const u16* __restrict__ hn,
    const float* __restrict__ w1, const float* __restrict__ b1,
    const float* __restrict__ w2, const float* __restrict__ b2,
    const float* __restrict__ w3, const float* __restrict__ b3,
    const float* __restrict__ wp, const float* __restrict__ bp,
    float* __restrict__ vout) {
  __shared__ float x0[CDIM];
  __shared__ float x1[NVH];
  __shared__ float x2[NVH];
  int tid = threadIdx.x;
  for (int b = 0; b < BNUM; ++b) {
    const u16* xr = hn + (size_t)b * TQ * CDIM;
    for (int c = tid; c < CDIM; c += 128) x0[c] = bf2f(xr[c]);
    __syncthreads();
    float a = b1[tid];
    for (int c = 0; c < CDIM; ++c) a += x0[c] * w1[c * NVH + tid];
    x1[tid] = gelu_f(a); __syncthreads();
    float a2 = b2[tid];
    for (int c = 0; c < NVH; ++c) a2 += x1[c] * w2[c * NVH + tid];
    x2[tid] = gelu_f(a2); __syncthreads();
    float a3 = b3[tid];
    for (int c = 0; c < NVH; ++c) a3 += x2[c] * w3[c * NVH + tid];
    __syncthreads();
    x1[tid] = gelu_f(a3); __syncthreads();
    if (tid == 0) {
      float p = bp[0];
      for (int c = 0; c < NVH; ++c) p += x1[c] * wp[c];
      vout[b] = p;
    }
    __syncthreads();
  }
}

// ---------------------------------------------------------------------------
// Host orchestration
// ---------------------------------------------------------------------------
static inline void conv_bf16(hipStream_t s, const float* in, u16* out, long n) {
  long blocks = (n + 255) / 256; if (blocks > 16384) blocks = 16384;
  f32_to_bf16_kernel<<<dim3((unsigned)blocks), dim3(256), 0, s>>>(in, out, n);
}
static inline void launch_tr(hipStream_t s, const float* in, u16* out,
                             int rows, int cols, int ldi,
                             long sIb, long sIh, int Hdim, int Z) {
  dim3 grid(cols / 32, rows / 32, Z);
  transpose_f32_bf16_kernel<<<grid, dim3(256), 0, s>>>(in, out, rows, cols, ldi,
                                                       sIb, sIh, Hdim);
}
static inline void launch_gemm(hipStream_t s, const u16* A, const u16* B,
                               const float* bias, float* C,
                               int M, int N, int K, int lda, int ldb, int ldc,
                               long sAb, long sAh, long sBb, long sBh,
                               long sCb, long sCh, int Hdim, int Z,
                               float alpha) {
  dim3 grid((N + TN - 1) / TN, (M + TM - 1) / TM, Z);
  gemm_wmma_kernel<<<grid, dim3(256), 0, s>>>(A, B, bias, C, M, N, K, lda, ldb,
                                              ldc, sAb, sAh, sBb, sBh, sCb, sCh,
                                              Hdim, alpha);
}

extern "C" void kernel_launch(void* const* d_in, const int* in_sizes, int n_in,
                              void* d_out, int out_size, void* d_ws, size_t ws_size,
                              hipStream_t stream) {
  // ---- inputs, in setup_inputs() dict insertion order ----
  const int*   x     = (const int*)d_in[0];
  const float* y     = (const float*)d_in[1];
  const float* wte   = (const float*)d_in[2];
  const float* wpe   = (const float*)d_in[3];
  const float* ln1_s = (const float*)d_in[4];
  const float* ln1_b = (const float*)d_in[5];
  const float* attn_w  = (const float*)d_in[6];
  const float* attn_b  = (const float*)d_in[7];
  const float* attn_pw = (const float*)d_in[8];
  const float* attn_pb = (const float*)d_in[9];
  const float* ln2_s = (const float*)d_in[10];
  const float* ln2_b = (const float*)d_in[11];
  const float* xq_w  = (const float*)d_in[12];
  const float* xq_b  = (const float*)d_in[13];
  const float* ykv_w = (const float*)d_in[14];
  const float* ykv_b = (const float*)d_in[15];
  const float* cpw   = (const float*)d_in[16];
  const float* cpb   = (const float*)d_in[17];
  const float* ln3_s = (const float*)d_in[18];
  const float* ln3_b = (const float*)d_in[19];
  const float* fc_w  = (const float*)d_in[20];
  const float* fc_b  = (const float*)d_in[21];
  const float* fpw   = (const float*)d_in[22];
  const float* fpb   = (const float*)d_in[23];
  const float* lnf_s = (const float*)d_in[24];
  const float* lnf_b = (const float*)d_in[25];
  const float* vh_w1 = (const float*)d_in[26];
  const float* vh_b1 = (const float*)d_in[27];
  const float* vh_w2 = (const float*)d_in[28];
  const float* vh_b2 = (const float*)d_in[29];
  const float* vh_w3 = (const float*)d_in[30];
  const float* vh_b3 = (const float*)d_in[31];
  const float* vh_wp = (const float*)d_in[32];
  const float* vh_bp = (const float*)d_in[33];
  (void)in_sizes; (void)n_in; (void)out_size; (void)ws_size;
  float* out = (float*)d_out;

  // ---- workspace carve-up ----
  char* w = (char*)d_ws;
  auto alloc = [&](size_t bytes) {
    char* p = w; w += (bytes + 255) & ~(size_t)255; return p;
  };
  float* h      = (float*)alloc((size_t)NTOK * CDIM * 4);
  u16*   hn_bf  = (u16*)  alloc((size_t)NTOK * CDIM * 2);
  float* big    = (float*)alloc((size_t)NTOK * 3072 * 4);
  u16*   big_bf = (u16*)  alloc((size_t)NTOK * 3072 * 2);
  float* t1     = (float*)alloc((size_t)NTOK * CDIM * 4);
  u16*   o_bf   = (u16*)  alloc((size_t)NTOK * CDIM * 2);
  float* scores = (float*)alloc((size_t)BNUM * HNUM * TQ * TQ * 4);
  u16*   probs  = (u16*)  alloc((size_t)BNUM * HNUM * TQ * TQ * 2);
  u16*   y_bf   = (u16*)  alloc((size_t)NTOK * CDIM * 2);
  u16*   w_bf   = (u16*)  alloc((size_t)CDIM * 3072 * 2);
  // V^T slab (96 x [64 x 512]) + slack so the N=64 GEMM's 128-wide tile
  // over-reads stay inside the allocation (garbage cols are never stored).
  u16*   vt     = (u16*)  alloc(((size_t)BNUM * HNUM * HD * TQ + (size_t)TN * TQ) * 2);

  const long nC   = (long)NTOK * CDIM;
  const long SATT = (long)TQ * TQ;
  const long SVT  = (long)HD * TQ;                      // 32768

  embed_kernel<<<dim3(NTOK), dim3(256), 0, stream>>>(x, wte, wpe, h);
  conv_bf16(stream, y, y_bf, nC);

  for (int i = 0; i < LNUM; ++i) {
    const float* l1s = ln1_s + (size_t)i * CDIM;
    const float* l1b = ln1_b + (size_t)i * CDIM;
    const float* aw  = attn_w  + (size_t)i * CDIM * 2304;
    const float* ab  = attn_b  + (size_t)i * 2304;
    const float* apw = attn_pw + (size_t)i * CDIM * CDIM;
    const float* apb = attn_pb + (size_t)i * CDIM;
    const float* l2s = ln2_s + (size_t)i * CDIM;
    const float* l2b = ln2_b + (size_t)i * CDIM;
    const float* xqw = xq_w  + (size_t)i * CDIM * CDIM;
    const float* xqb = xq_b  + (size_t)i * CDIM;
    const float* kvw = ykv_w + (size_t)i * CDIM * 1536;
    const float* kvb = ykv_b + (size_t)i * 1536;
    const float* cw  = cpw   + (size_t)i * CDIM * CDIM;
    const float* cb  = cpb   + (size_t)i * CDIM;
    const float* l3s = ln3_s + (size_t)i * CDIM;
    const float* l3b = ln3_b + (size_t)i * CDIM;
    const float* fw  = fc_w  + (size_t)i * CDIM * 3072;
    const float* fb  = fc_b  + (size_t)i * 3072;
    const float* pw  = fpw   + (size_t)i * 3072 * CDIM;
    const float* pb  = fpb   + (size_t)i * CDIM;

    // ---------------- self-attention ----------------
    ln_bf16_kernel<<<dim3(NTOK), dim3(256), 0, stream>>>(h, l1s, l1b, hn_bf);
    launch_tr(stream, aw, w_bf, CDIM, 2304, 2304, 0, 0, 1, 1);        // W^T
    launch_gemm(stream, hn_bf, w_bf, ab, big, NTOK, 2304, CDIM,
                CDIM, CDIM, 2304, 0, 0, 0, 0, 0, 0, 1, 1, 1.0f);      // qkv
    conv_bf16(stream, big, big_bf, (long)NTOK * 2304);
    // V^T per (b,h): vt[z][d][t] = qkv[b, t, 1536 + h*64 + d]
    launch_tr(stream, big + 1536, vt, TQ, HD, 2304,
              (long)TQ * 2304, HD, HNUM, BNUM * HNUM);
    // scores = q @ k^T / 8
    launch_gemm(stream, big_bf, big_bf + 768, nullptr, scores,
                TQ, TQ, HD, 2304, 2304, TQ,
                (long)TQ * 2304, HD, (long)TQ * 2304, HD,
                (long)HNUM * SATT, SATT, HNUM, BNUM * HNUM, 0.125f);
    softmax_bf16_kernel<<<dim3(BNUM * HNUM * TQ), dim3(256), 0, stream>>>(scores, probs);
    // o = probs @ v  (B = V^T, contiguous-k) -> (B,T,C) head-interleaved
    launch_gemm(stream, probs, vt, nullptr, t1,
                TQ, HD, TQ, TQ, TQ, CDIM,
                (long)HNUM * SATT, SATT, (long)HNUM * SVT, SVT,
                (long)TQ * CDIM, HD, HNUM, BNUM * HNUM, 1.0f);
    conv_bf16(stream, t1, o_bf, nC);
    launch_tr(stream, apw, w_bf, CDIM, CDIM, CDIM, 0, 0, 1, 1);
    launch_gemm(stream, o_bf, w_bf, apb, t1, NTOK, CDIM, CDIM,
                CDIM, CDIM, CDIM, 0, 0, 0, 0, 0, 0, 1, 1, 1.0f);
    add_inplace_kernel<<<dim3(8192), dim3(256), 0, stream>>>(h, t1, nC);

    // ---------------- cross-attention ----------------
    ln_bf16_kernel<<<dim3(NTOK), dim3(256), 0, stream>>>(h, l2s, l2b, hn_bf);
    launch_tr(stream, xqw, w_bf, CDIM, CDIM, CDIM, 0, 0, 1, 1);
    launch_gemm(stream, hn_bf, w_bf, xqb, t1, NTOK, CDIM, CDIM,
                CDIM, CDIM, CDIM, 0, 0, 0, 0, 0, 0, 1, 1, 1.0f);      // q
    conv_bf16(stream, t1, o_bf, nC);
    launch_tr(stream, kvw, w_bf, CDIM, 1536, 1536, 0, 0, 1, 1);
    launch_gemm(stream, y_bf, w_bf, kvb, big, NTOK, 1536, CDIM,
                CDIM, CDIM, 1536, 0, 0, 0, 0, 0, 0, 1, 1, 1.0f);      // kv
    conv_bf16(stream, big, big_bf, (long)NTOK * 1536);
    launch_tr(stream, big + 768, vt, TQ, HD, 1536,
              (long)TQ * 1536, HD, HNUM, BNUM * HNUM);                // V^T
    launch_gemm(stream, o_bf, big_bf, nullptr, scores,
                TQ, TQ, HD, CDIM, 1536, TQ,
                (long)TQ * CDIM, HD, (long)TQ * 1536, HD,
                (long)HNUM * SATT, SATT, HNUM, BNUM * HNUM, 0.125f);
    softmax_bf16_kernel<<<dim3(BNUM * HNUM * TQ), dim3(256), 0, stream>>>(scores, probs);
    launch_gemm(stream, probs, vt, nullptr, t1,
                TQ, HD, TQ, TQ, TQ, CDIM,
                (long)HNUM * SATT, SATT, (long)HNUM * SVT, SVT,
                (long)TQ * CDIM, HD, HNUM, BNUM * HNUM, 1.0f);
    conv_bf16(stream, t1, o_bf, nC);
    launch_tr(stream, cw, w_bf, CDIM, CDIM, CDIM, 0, 0, 1, 1);
    launch_gemm(stream, o_bf, w_bf, cb, t1, NTOK, CDIM, CDIM,
                CDIM, CDIM, CDIM, 0, 0, 0, 0, 0, 0, 1, 1, 1.0f);
    add_inplace_kernel<<<dim3(8192), dim3(256), 0, stream>>>(h, t1, nC);

    // ---------------- MLP ----------------
    ln_bf16_kernel<<<dim3(NTOK), dim3(256), 0, stream>>>(h, l3s, l3b, hn_bf);
    launch_tr(stream, fw, w_bf, CDIM, 3072, 3072, 0, 0, 1, 1);
    launch_gemm(stream, hn_bf, w_bf, fb, big, NTOK, 3072, CDIM,
                CDIM, CDIM, 3072, 0, 0, 0, 0, 0, 0, 1, 1, 1.0f);
    {
      long n = (long)NTOK * 3072;
      long blocks = (n + 255) / 256; if (blocks > 16384) blocks = 16384;
      gelu_bf16_kernel<<<dim3((unsigned)blocks), dim3(256), 0, stream>>>(big, big_bf, n);
    }
    launch_tr(stream, pw, w_bf, 3072, CDIM, CDIM, 0, 0, 1, 1);
    launch_gemm(stream, big_bf, w_bf, pb, t1, NTOK, CDIM, 3072,
                3072, 3072, CDIM, 0, 0, 0, 0, 0, 0, 1, 1, 1.0f);
    add_inplace_kernel<<<dim3(8192), dim3(256), 0, stream>>>(h, t1, nC);
  }

  // ---- final LN + heads ----
  ln_bf16_kernel<<<dim3(NTOK), dim3(256), 0, stream>>>(h, lnf_s, lnf_b, hn_bf);
  policy_kernel<<<dim3(NTOK), dim3(256), 0, stream>>>(hn_bf, wte, out);
  value_head_kernel<<<dim3(1), dim3(128), 0, stream>>>(
      hn_bf, vh_w1, vh_b1, vh_w2, vh_b2, vh_w3, vh_b3, vh_wp, vh_bp,
      out + (size_t)NTOK * VOC);
}
